// IEConvLayer_42116449305122
// MI455X (gfx1250) — matrix operand order repaired and checked by reading, not compiled, not run on hardware
//
#include <hip/hip_runtime.h>

#define N_NODES 10000
#define N_EDGES 320000
#define IN_D    128
#define HID     32
#define OUT_D   128
#define KH      32
#define BN_EPS  1e-5f

typedef __attribute__((ext_vector_type(16))) __bf16 v16bf;
typedef __attribute__((ext_vector_type(8)))  float  v8f;

// ---------------------------------------------------------------------------
// Generic column stats: one block per column, biased variance (torch BN).
// ---------------------------------------------------------------------------
__global__ void col_stats_kernel(const float* __restrict__ x, int rows, int cols,
                                 float* __restrict__ mean, float* __restrict__ var) {
    __shared__ float s_sum[256];
    __shared__ float s_sq[256];
    const int c = blockIdx.x;
    float sum = 0.f, sq = 0.f;
    for (int r = threadIdx.x; r < rows; r += 256) {
        float v = x[(size_t)r * cols + c];
        sum += v; sq += v * v;
    }
    s_sum[threadIdx.x] = sum;
    s_sq[threadIdx.x]  = sq;
    __syncthreads();
    for (int s = 128; s > 0; s >>= 1) {
        if (threadIdx.x < s) {
            s_sum[threadIdx.x] += s_sum[threadIdx.x + s];
            s_sq[threadIdx.x]  += s_sq[threadIdx.x + s];
        }
        __syncthreads();
    }
    if (threadIdx.x == 0) {
        float m = s_sum[0] / (float)rows;
        mean[c] = m;
        var[c]  = s_sq[0] / (float)rows - m * m;
    }
}

// ---------------------------------------------------------------------------
// Hn = bf16(relu(bn(H)))  (elementwise)
// ---------------------------------------------------------------------------
__global__ void bn_relu_bf16_kernel(const float* __restrict__ x,
                                    const float* __restrict__ mean, const float* __restrict__ var,
                                    const float* __restrict__ g, const float* __restrict__ beta,
                                    __bf16* __restrict__ y, int rows, int cols) {
    int idx = blockIdx.x * 256 + threadIdx.x;
    if (idx >= rows * cols) return;
    int c = idx % cols;
    float v = g[c] * (x[idx] - mean[c]) * rsqrtf(var[c] + BN_EPS) + beta[c];
    y[idx] = (__bf16)fmaxf(v, 0.f);
}

__global__ void zero_kernel(float* __restrict__ p, int n) {
    int i = blockIdx.x * 256 + threadIdx.x;
    if (i < n) p[i] = 0.f;
}

// ---------------------------------------------------------------------------
// B-fragment pre-swizzle kernels: per-lane WMMA B register order
// ([tile][lane][16 bf16]): lane = (n%16) + 16*hi, element i -> k = i + hi*16.
// Each lane then fetches its fragment as one aligned 32B contiguous load.
// ---------------------------------------------------------------------------
__global__ void pack_w1_kernel(const float* __restrict__ w1, __bf16* __restrict__ frag) {
    int idx = blockIdx.x * 256 + threadIdx.x;      // 4 chunks * 2 tiles * 32 lanes * 16
    if (idx >= 4 * 2 * 32 * 16) return;
    int i = idx & 15, lane = (idx >> 4) & 31, tile = (idx >> 9) & 1, chunk = idx >> 10;
    int n = tile * 16 + (lane & 15);
    int k = chunk * 32 + i + (lane >> 4) * 16;
    frag[idx] = (__bf16)w1[(size_t)n * IN_D + k];
}

// U[k, c] = kw2[(1+j)*32 + k, m],  c = j*32+m.  64 column tiles of 16.
__global__ void pack_u_kernel(const float* __restrict__ kw2, __bf16* __restrict__ frag) {
    int idx = blockIdx.x * 256 + threadIdx.x;      // 64 tiles * 32 lanes * 16
    if (idx >= 64 * 32 * 16) return;
    int i = idx & 15, lane = (idx >> 4) & 31, ct = idx >> 9;
    int c = ct * 16 + (lane & 15);
    int j = c >> 5, mm = c & 31;
    int k = i + (lane >> 4) * 16;
    frag[idx] = (__bf16)kw2[(size_t)((1 + j) * 32 + k) * KH + mm];
}

// B2[k, j] = kb2[(1+j)*32 + k]  (constant part of the dynamic kernel), 2 tiles.
__global__ void pack_kb2_kernel(const float* __restrict__ kb2, __bf16* __restrict__ frag) {
    int idx = blockIdx.x * 256 + threadIdx.x;      // 2 tiles * 32 lanes * 16
    if (idx >= 2 * 32 * 16) return;
    int i = idx & 15, lane = (idx >> 4) & 31, tile = idx >> 9;
    int j = tile * 16 + (lane & 15);
    int k = i + (lane >> 4) * 16;
    frag[idx] = (__bf16)kb2[(1 + j) * 32 + k];
}

// B3[m, j] = kw2[j, m]  (bias row of the dynamic kernel vs h_e), 2 tiles.
__global__ void pack_kw2b_kernel(const float* __restrict__ kw2, __bf16* __restrict__ frag) {
    int idx = blockIdx.x * 256 + threadIdx.x;      // 2 tiles * 32 lanes * 16
    if (idx >= 2 * 32 * 16) return;
    int i = idx & 15, lane = (idx >> 4) & 31, tile = idx >> 9;
    int j = tile * 16 + (lane & 15);
    int mloc = i + (lane >> 4) * 16;
    frag[idx] = (__bf16)kw2[(size_t)j * KH + mloc];
}

__global__ void pack_w2_kernel(const float* __restrict__ w2, __bf16* __restrict__ frag) {
    int idx = blockIdx.x * 256 + threadIdx.x;      // 8 tiles * 32 lanes * 16
    if (idx >= 8 * 32 * 16) return;
    int i = idx & 15, lane = (idx >> 4) & 31, tile = idx >> 9;
    int n = tile * 16 + (lane & 15);
    int k = i + (lane >> 4) * 16;
    frag[idx] = (__bf16)w2[(size_t)n * HID + k];
}

// ---------------------------------------------------------------------------
// msg_raw[E,32] = gather(Hn, node_in) @ w1^T + b1
// One wave per 16 edges; pre-swizzled B fragments, 8 WMMAs / block.
// ---------------------------------------------------------------------------
__global__ void msg_gemm_kernel(const __bf16* __restrict__ Hn,
                                const int* __restrict__ node_in,
                                const v16bf* __restrict__ w1frag, const float* __restrict__ b1,
                                float* __restrict__ msg_raw) {
    const int base = blockIdx.x * 16;
    const int lane = threadIdx.x & 31;
    const int m    = lane & 15;
    const int hi   = lane >> 4;
    const int node = node_in[base + m];
    const __bf16* row = Hn + (size_t)node * IN_D;

    v8f c0 = {}; v8f c1 = {};
    #pragma unroll
    for (int chunk = 0; chunk < 4; ++chunk) {
        v16bf a;
        #pragma unroll
        for (int i = 0; i < 16; ++i) {
            int k = chunk * 32 + i + hi * 16;
            a[i] = row[k];
        }
        v16bf b0  = w1frag[(chunk * 2 + 0) * 32 + lane];
        v16bf b1v = w1frag[(chunk * 2 + 1) * 32 + lane];
        c0 = __builtin_amdgcn_wmma_f32_16x16x32_bf16(false, a, false, b0,  (short)0, c0, false, false);
        c1 = __builtin_amdgcn_wmma_f32_16x16x32_bf16(false, a, false, b1v, (short)0, c1, false, false);
    }
    #pragma unroll
    for (int r = 0; r < 8; ++r) {
        int e = base + r + 8 * hi;
        msg_raw[(size_t)e * HID + m]      = c0[r] + b1[m];
        msg_raw[(size_t)e * HID + m + 16] = c1[r] + b1[m + 16];
    }
}

// ---------------------------------------------------------------------------
// Fused dynamic-kernel edge pass.  Per block: 16 edges, 256 threads (8 waves).
//   msgN = relu(bn(msg_raw)), h_e = relu(edge_attr*kw1+kb1)      -> LDS
//   sConst = msgN @ B2 (waves 0-1), sBias = h @ B3 (waves 2-3)   -> LDS
//   S      = msgN @ U  (two 512-col passes; 8 waves x 4 WMMA)    -> LDS
//   msg_new[e,j] = kb2[j] + sBias[e][j] + sConst[e][j] + S[e,j*32+:]·h_e
//   atomic scatter-add into upd[node_out[e], j]
// ---------------------------------------------------------------------------
__global__ void edge_kernel(const float* __restrict__ msg_raw,
                            const float* __restrict__ edge_attr,
                            const int*   __restrict__ node_out,
                            const float* __restrict__ kw1, const float* __restrict__ kb1,
                            const float* __restrict__ kb2,
                            const v16bf* __restrict__ ufrag,
                            const v16bf* __restrict__ kb2frag,
                            const v16bf* __restrict__ kw2bfrag,
                            const float* __restrict__ mmean, const float* __restrict__ mvar,
                            const float* __restrict__ g_msg, const float* __restrict__ beta_msg,
                            float* __restrict__ upd) {
    __shared__ float sS[16][512];
    __shared__ float sMsg[16][HID];
    __shared__ float sH[16][KH];
    __shared__ float sConst[16][HID];
    __shared__ float sBias[16][HID];

    const int base = blockIdx.x * 16;
    const int t    = threadIdx.x;

    for (int i = t; i < 16 * HID; i += 256) {
        int e = i >> 5, k = i & 31;
        float v = msg_raw[(size_t)(base + e) * HID + k];
        v = g_msg[k] * (v - mmean[k]) * rsqrtf(mvar[k] + BN_EPS) + beta_msg[k];
        sMsg[e][k] = fmaxf(v, 0.f);
        float hv = edge_attr[base + e] * kw1[k] + kb1[k];
        sH[e][k] = fmaxf(hv, 0.f);
    }
    __syncthreads();

    const int lane = t & 31, wave = t >> 5;
    const int m = lane & 15, hi = lane >> 4;

    // A fragments [16 x K=32] from msgN and from h — built once per block.
    v16bf a, ah;
    #pragma unroll
    for (int i = 0; i < 16; ++i) {
        int k = i + hi * 16;
        a[i]  = (__bf16)sMsg[m][k];
        ah[i] = (__bf16)sH[m][k];
    }

    // Constant-kernel and bias-row GEMMs on the tensor path (waves 0-3).
    if (wave < 2) {
        v16bf b = kb2frag[wave * 32 + lane];
        v8f cc = {};
        cc = __builtin_amdgcn_wmma_f32_16x16x32_bf16(false, a, false, b, (short)0, cc, false, false);
        #pragma unroll
        for (int r = 0; r < 8; ++r)
            sConst[r + 8 * hi][wave * 16 + m] = cc[r];
    } else if (wave < 4) {
        v16bf b = kw2bfrag[(wave - 2) * 32 + lane];
        v8f cc = {};
        cc = __builtin_amdgcn_wmma_f32_16x16x32_bf16(false, ah, false, b, (short)0, cc, false, false);
        #pragma unroll
        for (int r = 0; r < 8; ++r)
            sBias[r + 8 * hi][(wave - 2) * 16 + m] = cc[r];
    }

    for (int pass = 0; pass < 2; ++pass) {
        #pragma unroll
        for (int q = 0; q < 4; ++q) {
            int ct = wave * 4 + q;                         // tile within pass
            v16bf b = ufrag[(size_t)(pass * 32 + ct) * 32 + lane];
            v8f cacc = {};
            cacc = __builtin_amdgcn_wmma_f32_16x16x32_bf16(false, a, false, b, (short)0, cacc, false, false);
            #pragma unroll
            for (int r = 0; r < 8; ++r)
                sS[r + 8 * hi][ct * 16 + m] = cacc[r];
        }
        __syncthreads();

        {
            int e  = t >> 4;
            int jj = t & 15;
            int j  = pass * 16 + jj;
            float val = kb2[j] + sConst[e][j] + sBias[e][j];
            #pragma unroll 4
            for (int mm = 0; mm < HID; ++mm)       // dynamic part via S tile
                val += sS[e][jj * 32 + mm] * sH[e][mm];
            atomicAdd(&upd[(size_t)node_out[base + e] * HID + j], val);
        }
        __syncthreads();
    }
}

// ---------------------------------------------------------------------------
// out_raw[N,128] = relu(bn(upd)) @ w2^T + b2
// Per block: 16 nodes, 8 waves; wave w -> output column tile w.
// ---------------------------------------------------------------------------
__global__ void out_gemm_kernel(const float* __restrict__ upd,
                                const float* __restrict__ umean, const float* __restrict__ uvar,
                                const float* __restrict__ g_upd, const float* __restrict__ beta_upd,
                                const v16bf* __restrict__ w2frag, const float* __restrict__ b2,
                                float* __restrict__ out_raw) {
    __shared__ float sU[16][HID];
    const int base = blockIdx.x * 16;
    const int t = threadIdx.x;
    for (int i = t; i < 16 * HID; i += 256) {
        int r = i >> 5, k = i & 31;
        float v = upd[(size_t)(base + r) * HID + k];
        v = g_upd[k] * (v - umean[k]) * rsqrtf(uvar[k] + BN_EPS) + beta_upd[k];
        sU[r][k] = fmaxf(v, 0.f);
    }
    __syncthreads();

    const int lane = t & 31, wave = t >> 5;
    const int m = lane & 15, hi = lane >> 4;

    v16bf a;
    #pragma unroll
    for (int i = 0; i < 16; ++i) {
        int k = i + hi * 16;
        a[i] = (__bf16)sU[m][k];
    }
    v16bf b = w2frag[wave * 32 + lane];
    v8f c = {};
    c = __builtin_amdgcn_wmma_f32_16x16x32_bf16(false, a, false, b, (short)0, c, false, false);
    const int n = wave * 16 + m;
    #pragma unroll
    for (int r = 0; r < 8; ++r)
        out_raw[(size_t)(base + r + 8 * hi) * OUT_D + n] = c[r] + b2[n];
}

// ---------------------------------------------------------------------------
// d_out = bn(out_raw) (affine)
// ---------------------------------------------------------------------------
__global__ void bn_out_kernel(const float* __restrict__ x,
                              const float* __restrict__ mean, const float* __restrict__ var,
                              const float* __restrict__ g, const float* __restrict__ beta,
                              float* __restrict__ y, int rows, int cols) {
    int idx = blockIdx.x * 256 + threadIdx.x;
    if (idx >= rows * cols) return;
    int c = idx % cols;
    y[idx] = g[c] * (x[idx] - mean[c]) * rsqrtf(var[c] + BN_EPS) + beta[c];
}

// ---------------------------------------------------------------------------
extern "C" void kernel_launch(void* const* d_in, const int* in_sizes, int n_in,
                              void* d_out, int out_size, void* d_ws, size_t ws_size,
                              hipStream_t stream) {
    const float* H         = (const float*)d_in[0];
    const float* edge_attr = (const float*)d_in[1];
    const int*   edges     = (const int*)d_in[2];
    const int*   node_out  = edges;            // edges[0,:]
    const int*   node_in   = edges + N_EDGES;  // edges[1,:]
    const float* w1   = (const float*)d_in[3];
    const float* b1   = (const float*)d_in[4];
    const float* kw1  = (const float*)d_in[5];
    const float* kb1  = (const float*)d_in[6];
    const float* kw2  = (const float*)d_in[7];
    const float* kb2  = (const float*)d_in[8];
    const float* w2   = (const float*)d_in[9];
    const float* b2   = (const float*)d_in[10];
    const float* g_in     = (const float*)d_in[11];
    const float* beta_in  = (const float*)d_in[12];
    const float* g_msg    = (const float*)d_in[13];
    const float* beta_msg = (const float*)d_in[14];
    const float* g_upd    = (const float*)d_in[15];
    const float* beta_upd = (const float*)d_in[16];
    const float* g_out    = (const float*)d_in[17];
    const float* beta_out = (const float*)d_in[18];

    // workspace carve-up (256B aligned)
    char* ws = (char*)d_ws;
    size_t off = 0;
    auto alloc = [&](size_t bytes) -> void* {
        void* p = ws + off;
        off += (bytes + 255) & ~(size_t)255;
        return p;
    };
    float*  meanH    = (float*)alloc(IN_D * sizeof(float));
    float*  varH     = (float*)alloc(IN_D * sizeof(float));
    float*  mean_msg = (float*)alloc(HID * sizeof(float));
    float*  var_msg  = (float*)alloc(HID * sizeof(float));
    float*  mean_upd = (float*)alloc(HID * sizeof(float));
    float*  var_upd  = (float*)alloc(HID * sizeof(float));
    float*  mean_out = (float*)alloc(OUT_D * sizeof(float));
    float*  var_out  = (float*)alloc(OUT_D * sizeof(float));
    __bf16* w1frag   = (__bf16*)alloc((size_t)4 * 2 * 32 * 16 * sizeof(__bf16));
    __bf16* ufrag    = (__bf16*)alloc((size_t)64 * 32 * 16 * sizeof(__bf16));
    __bf16* kb2frag  = (__bf16*)alloc((size_t)2 * 32 * 16 * sizeof(__bf16));
    __bf16* kw2bfrag = (__bf16*)alloc((size_t)2 * 32 * 16 * sizeof(__bf16));
    __bf16* w2frag   = (__bf16*)alloc((size_t)8 * 32 * 16 * sizeof(__bf16));
    __bf16* Hn       = (__bf16*)alloc((size_t)N_NODES * IN_D * sizeof(__bf16));
    float*  msg_raw  = (float*)alloc((size_t)N_EDGES * HID * sizeof(float));
    float*  upd      = (float*)alloc((size_t)N_NODES * HID * sizeof(float));
    float*  out_raw  = (float*)alloc((size_t)N_NODES * OUT_D * sizeof(float));

    // 0) pre-swizzle static B operands into WMMA fragment order
    pack_w1_kernel<<<(4 * 2 * 32 * 16 + 255) / 256, 256, 0, stream>>>(w1, w1frag);
    pack_u_kernel<<<(64 * 32 * 16 + 255) / 256, 256, 0, stream>>>(kw2, ufrag);
    pack_kb2_kernel<<<(2 * 32 * 16 + 255) / 256, 256, 0, stream>>>(kb2, kb2frag);
    pack_kw2b_kernel<<<(2 * 32 * 16 + 255) / 256, 256, 0, stream>>>(kw2, kw2bfrag);
    pack_w2_kernel<<<(8 * 32 * 16 + 255) / 256, 256, 0, stream>>>(w2, w2frag);

    // 1) input BN stats + normalize/relu -> bf16
    col_stats_kernel<<<IN_D, 256, 0, stream>>>(H, N_NODES, IN_D, meanH, varH);
    bn_relu_bf16_kernel<<<(N_NODES * IN_D + 255) / 256, 256, 0, stream>>>(
        H, meanH, varH, g_in, beta_in, Hn, N_NODES, IN_D);

    // 2) message GEMM (gather + WMMA)
    msg_gemm_kernel<<<N_EDGES / 16, 32, 0, stream>>>(
        Hn, node_in, (const v16bf*)w1frag, b1, msg_raw);

    // 3) msg BN stats (msg_raw fits in L2)
    col_stats_kernel<<<HID, 256, 0, stream>>>(msg_raw, N_EDGES, HID, mean_msg, var_msg);

    // 4) zero aggregation target, then fused dynamic-kernel edge pass
    zero_kernel<<<(N_NODES * HID + 255) / 256, 256, 0, stream>>>(upd, N_NODES * HID);
    edge_kernel<<<N_EDGES / 16, 256, 0, stream>>>(
        msg_raw, edge_attr, node_out, kw1, kb1, kb2,
        (const v16bf*)ufrag, (const v16bf*)kb2frag, (const v16bf*)kw2bfrag,
        mean_msg, var_msg, g_msg, beta_msg, upd);

    // 5) upd BN stats, output GEMM
    col_stats_kernel<<<HID, 256, 0, stream>>>(upd, N_NODES, HID, mean_upd, var_upd);
    out_gemm_kernel<<<N_NODES / 16, 256, 0, stream>>>(
        upd, mean_upd, var_upd, g_upd, beta_upd, (const v16bf*)w2frag, b2, out_raw);

    // 6) output BN
    col_stats_kernel<<<OUT_D, 256, 0, stream>>>(out_raw, N_NODES, OUT_D, mean_out, var_out);
    bn_out_kernel<<<(N_NODES * OUT_D + 255) / 256, 256, 0, stream>>>(
        out_raw, mean_out, var_out, g_out, beta_out, (float*)d_out, N_NODES, OUT_D);
}
